// AutoEncoderTopK_48550310314117
// MI455X (gfx1250) — matrix operand order repaired and checked by reading, not compiled, not run on hardware
//
#include <hip/hip_runtime.h>
#include <stdint.h>

// Problem dims (fixed by reference setup_inputs).
#define BATCH 4096
#define D_IN  2048
#define D_SAE 16384
#define KTOP  100

// GEMM tiling.
#define BM 64
#define BN 128
#define BK 32
#define LDA 40                    // padded LDS row stride in bf16 elems (80B, 16B-aligned quads)
#define KTILES (D_IN / BK)        // 64

typedef __attribute__((ext_vector_type(16))) __bf16 v16bf;
typedef __attribute__((ext_vector_type(8)))  float  v8f;

union FragBF { v16bf v; uint4 q[2]; };

__device__ __forceinline__ uint32_t f32_bits(float f) {
  union { float f; uint32_t u; } c; c.f = f; return c.u;
}
__device__ __forceinline__ uint16_t bf16_rne(float f) {
  uint32_t u = f32_bits(f);
  u += 0x7FFFu + ((u >> 16) & 1u);       // round-to-nearest-even to bf16
  return (uint16_t)(u >> 16);
}
__device__ __forceinline__ float bf16_f(uint16_t h) {
  union { uint32_t u; float f; } c; c.u = ((uint32_t)h) << 16; return c.f;
}

// Raw 32-bit LDS offset of a __shared__ object (async-to-LDS VDST wants the
// wave-relative LDS byte address, not a generic pointer).
typedef __attribute__((address_space(3))) uint16_t lds_u16;
__device__ __forceinline__ unsigned lds_off16(uint16_t* p) {
  return (unsigned)(size_t)(lds_u16*)p;
}

// CDNA5 async copy: global -> LDS, 16B per lane, tracked by ASYNCcnt.
__device__ __forceinline__ void async_copy_b128(uint16_t* lds_dst, const uint16_t* gsrc) {
  unsigned l = lds_off16(lds_dst);
  asm volatile("global_load_async_to_lds_b128 %0, %1, off"
               :: "v"(l), "v"(gsrc) : "memory");
}
__device__ __forceinline__ void wait_async0() {
  asm volatile("s_wait_asynccnt 0x0" ::: "memory");
}

// ---------------------------------------------------------------------------
// Prep: A = x - b_dec  ->  bf16 hi/lo planes, row-major [BATCH][D_IN]
// ---------------------------------------------------------------------------
__global__ __launch_bounds__(256) void prep_x_kernel(
    const float* __restrict__ x, const float* __restrict__ b_dec,
    uint16_t* __restrict__ Ahi, uint16_t* __restrict__ Alo)
{
  const size_t gid  = (size_t)blockIdx.x * 256 + threadIdx.x;
  const size_t base = gid * 4;
  if (base >= (size_t)BATCH * D_IN) return;
  const int col = (int)(base & (D_IN - 1));
  float4 xv = *(const float4*)(x + base);
  float4 bv = *(const float4*)(b_dec + col);
  float a[4] = {xv.x - bv.x, xv.y - bv.y, xv.z - bv.z, xv.w - bv.w};
  uint16_t h[4], l[4];
#pragma unroll
  for (int i = 0; i < 4; ++i) {
    h[i] = bf16_rne(a[i]);
    l[i] = bf16_rne(a[i] - bf16_f(h[i]));
  }
  uint2 hp = make_uint2((uint32_t)h[0] | ((uint32_t)h[1] << 16),
                        (uint32_t)h[2] | ((uint32_t)h[3] << 16));
  uint2 lp = make_uint2((uint32_t)l[0] | ((uint32_t)l[1] << 16),
                        (uint32_t)l[2] | ((uint32_t)l[3] << 16));
  *(uint2*)(Ahi + base) = hp;
  *(uint2*)(Alo + base) = lp;
}

// ---------------------------------------------------------------------------
// Prep: W_enc [D_IN][D_SAE] -> transposed bf16 hi/lo planes [D_SAE][D_IN]
// (K-contiguous rows so B fragments load as two ds_load_b128)
// ---------------------------------------------------------------------------
__global__ __launch_bounds__(256) void prep_w_kernel(
    const float* __restrict__ W, uint16_t* __restrict__ Whi, uint16_t* __restrict__ Wlo)
{
  __shared__ float tile[32][33];
  const int n0 = blockIdx.x * 32;
  const int k0 = blockIdx.y * 32;
  const int tx = threadIdx.x;   // 0..31
  const int ty = threadIdx.y;   // 0..7
#pragma unroll
  for (int i = 0; i < 4; ++i) {
    int k = k0 + ty + i * 8;
    tile[ty + i * 8][tx] = W[(size_t)k * D_SAE + n0 + tx];
  }
  __syncthreads();
#pragma unroll
  for (int i = 0; i < 4; ++i) {
    int n = n0 + ty + i * 8;
    float a = tile[tx][ty + i * 8];
    uint16_t h = bf16_rne(a);
    uint16_t l = bf16_rne(a - bf16_f(h));
    Whi[(size_t)n * D_IN + k0 + tx] = h;
    Wlo[(size_t)n * D_IN + k0 + tx] = l;
  }
}

// ---------------------------------------------------------------------------
// Encoder GEMM: post = relu(A @ W_enc + b_enc), split-bf16 (hi*hi+hi*lo+lo*hi)
// Block: 64x128 tile, 8 waves (2x4), each wave 32x32 via 2x2 WMMA tiles.
// Global->LDS staging uses CDNA5 async B128 copies (no staging VGPRs).
// ---------------------------------------------------------------------------
__global__ __launch_bounds__(256) void encode_gemm_kernel(
    const uint16_t* __restrict__ Ahi, const uint16_t* __restrict__ Alo,
    const uint16_t* __restrict__ Whi, const uint16_t* __restrict__ Wlo,
    const float* __restrict__ b_enc, float* __restrict__ post)
{
  __shared__ __align__(16) uint16_t sAh[2][BM * LDA];
  __shared__ __align__(16) uint16_t sAl[2][BM * LDA];
  __shared__ __align__(16) uint16_t sBh[2][BN * LDA];
  __shared__ __align__(16) uint16_t sBl[2][BN * LDA];

  const int tid   = threadIdx.x;
  const int lane  = tid & 31;
  const int wv    = tid >> 5;
  const int waveM = wv >> 2;        // 0..1
  const int waveN = wv & 3;         // 0..3
  const int m0 = blockIdx.y * BM;
  const int n0 = blockIdx.x * BN;

  // global->LDS staging indices: 16B (8 bf16) chunks per thread
  const int aRow = tid >> 2;
  const int aKc  = (tid & 3) * 8;
  const int bRow0 = tid >> 2;                 // chunk 0 row (0..63)
  const int bRow1 = (tid + 256) >> 2;         // chunk 1 row (64..127)

  // Issue one stage of async global->LDS copies (6 x B128 per thread).
  auto stage = [&](int kt, int buf) {
    const size_t kb   = (size_t)kt * BK;
    const size_t aoff = (size_t)(m0 + aRow) * D_IN + kb + aKc;
    async_copy_b128(&sAh[buf][aRow * LDA + aKc], Ahi + aoff);
    async_copy_b128(&sAl[buf][aRow * LDA + aKc], Alo + aoff);
    const size_t b0 = (size_t)(n0 + bRow0) * D_IN + kb + aKc;
    const size_t b1 = (size_t)(n0 + bRow1) * D_IN + kb + aKc;
    async_copy_b128(&sBh[buf][bRow0 * LDA + aKc], Whi + b0);
    async_copy_b128(&sBl[buf][bRow0 * LDA + aKc], Wlo + b0);
    async_copy_b128(&sBh[buf][bRow1 * LDA + aKc], Whi + b1);
    async_copy_b128(&sBl[buf][bRow1 * LDA + aKc], Wlo + b1);
  };

  v8f zero = {0.f, 0.f, 0.f, 0.f, 0.f, 0.f, 0.f, 0.f};
  v8f acc[2][2];
#pragma unroll
  for (int mi = 0; mi < 2; ++mi)
#pragma unroll
    for (int ni = 0; ni < 2; ++ni) acc[mi][ni] = zero;

  const int lr = lane & 15;
  const int hf = lane >> 4;

  stage(0, 0);
  wait_async0();
  __syncthreads();

  for (int kt = 0; kt < KTILES; ++kt) {
    const int buf = kt & 1;
    if (kt + 1 < KTILES) stage(kt + 1, buf ^ 1);
    if (kt + 2 < KTILES) {  // gfx1250 global_prefetch_b8 of next-next B tile
      __builtin_prefetch(Whi + (size_t)(n0 + (tid >> 1)) * D_IN + (size_t)(kt + 2) * BK
                             + (tid & 1) * 16, 0, 3);
    }

    // B fragments: 32x16 bf16, lanes 0-15 K=0..15, lanes 16-31 K=16..31 (contiguous k order, same as A)
    FragBF bh[2], bl[2];
#pragma unroll
    for (int ni = 0; ni < 2; ++ni) {
      const int bo = (waveN * 32 + ni * 16 + lr) * LDA + hf * 16;
      bh[ni].q[0] = *(const uint4*)&sBh[buf][bo];
      bh[ni].q[1] = *(const uint4*)&sBh[buf][bo + 8];
      bl[ni].q[0] = *(const uint4*)&sBl[buf][bo];
      bl[ni].q[1] = *(const uint4*)&sBl[buf][bo + 8];
    }
#pragma unroll
    for (int mi = 0; mi < 2; ++mi) {
      // A fragment: 16x32 bf16, kbase = (lane>=16)?16:0, quads at +0 / +8
      FragBF ah, al;
      const int ao = (waveM * 32 + mi * 16 + lr) * LDA + hf * 16;
      ah.q[0] = *(const uint4*)&sAh[buf][ao];
      ah.q[1] = *(const uint4*)&sAh[buf][ao + 8];
      al.q[0] = *(const uint4*)&sAl[buf][ao];
      al.q[1] = *(const uint4*)&sAl[buf][ao + 8];
#pragma unroll
      for (int ni = 0; ni < 2; ++ni) {
        acc[mi][ni] = __builtin_amdgcn_wmma_f32_16x16x32_bf16(
            false, ah.v, false, bh[ni].v, (short)0, acc[mi][ni], false, false);
        acc[mi][ni] = __builtin_amdgcn_wmma_f32_16x16x32_bf16(
            false, ah.v, false, bl[ni].v, (short)0, acc[mi][ni], false, false);
        acc[mi][ni] = __builtin_amdgcn_wmma_f32_16x16x32_bf16(
            false, al.v, false, bh[ni].v, (short)0, acc[mi][ni], false, false);
      }
    }

    wait_async0();      // own async writes to buf^1 landed in LDS
    __syncthreads();    // everyone's writes visible before next iteration reads
  }

  // Epilogue: bias + relu, C/D layout: VGPR i -> row i + 8*(lane>=16), col = lane&15
#pragma unroll
  for (int mi = 0; mi < 2; ++mi) {
#pragma unroll
    for (int ni = 0; ni < 2; ++ni) {
      const int gn   = n0 + waveN * 32 + ni * 16 + lr;
      const float bias = b_enc[gn];
#pragma unroll
      for (int i = 0; i < 8; ++i) {
        const int gm = m0 + waveM * 32 + mi * 16 + i + hf * 8;
        float v = acc[mi][ni][i] + bias;
        v = fmaxf(v, 0.0f);
        post[(size_t)gm * D_SAE + gn] = v;
      }
    }
  }
}

// ---------------------------------------------------------------------------
// Top-K per row via 2-level 2048-bin radix select on float bits (vals >= 0),
// then deterministic bitmap/popcount emission (ties -> lowest index).
// ---------------------------------------------------------------------------
__global__ __launch_bounds__(256) void topk_kernel(
    const float* __restrict__ post, float* __restrict__ vals, int* __restrict__ idxs)
{
  __shared__ unsigned int hist[2048];
  __shared__ unsigned int part[256];
  __shared__ unsigned int bm[512], tbm[512], pre[512];
  __shared__ unsigned int s_sel, s_above;

  const int row = blockIdx.x;
  const int tid = threadIdx.x;
  const float* p = post + (size_t)row * D_SAE;

  // ---- level 0: bins = bits[31:21]
  for (int b = tid; b < 2048; b += 256) hist[b] = 0;
  __syncthreads();
  for (int j = tid; j < D_SAE; j += 256)
    atomicAdd(&hist[f32_bits(p[j]) >> 21], 1u);
  __syncthreads();
  unsigned int c8 = 0;
#pragma unroll
  for (int i = 0; i < 8; ++i) c8 += hist[tid * 8 + i];
  part[tid] = c8;
  __syncthreads();
  if (tid == 0) {
    unsigned int c = 0; int ch = 255;
    while (ch > 0 && c + part[ch] < KTOP) { c += part[ch]; --ch; }
    int b = ch * 8 + 7;
    while (b > ch * 8 && c + hist[b] < KTOP) { c += hist[b]; --b; }
    s_sel = (unsigned int)b; s_above = c;
  }
  __syncthreads();
  const unsigned int sel0   = s_sel;
  const unsigned int above0 = s_above;
  const unsigned int need1  = KTOP - above0;          // >= 1
  __syncthreads();

  // ---- level 1: refine within bin sel0, bins = bits[20:10]
  for (int b = tid; b < 2048; b += 256) hist[b] = 0;
  __syncthreads();
  for (int j = tid; j < D_SAE; j += 256) {
    uint32_t u = f32_bits(p[j]);
    if ((u >> 21) == sel0) atomicAdd(&hist[(u >> 10) & 0x7FFu], 1u);
  }
  __syncthreads();
  c8 = 0;
#pragma unroll
  for (int i = 0; i < 8; ++i) c8 += hist[tid * 8 + i];
  part[tid] = c8;
  __syncthreads();
  if (tid == 0) {
    unsigned int c = 0; int ch = 255;
    while (ch > 0 && c + part[ch] < need1) { c += part[ch]; --ch; }
    int b = ch * 8 + 7;
    while (b > ch * 8 && c + hist[b] < need1) { c += hist[b]; --b; }
    s_sel = (unsigned int)b; s_above = c;
  }
  __syncthreads();
  const unsigned int sel1   = s_sel;
  const unsigned int above1 = s_above;
  const unsigned int nTie   = need1 - above1;         // >= 1

  // ---- deterministic selection bitmaps
  for (int w = tid; w < 512; w += 256) { bm[w] = 0; tbm[w] = 0; }
  __syncthreads();
  for (int j = tid; j < D_SAE; j += 256) {
    uint32_t u  = f32_bits(p[j]);
    uint32_t k0 = u >> 21;
    if (k0 > sel0) {
      atomicOr(&bm[j >> 5], 1u << (j & 31));
    } else if (k0 == sel0) {
      uint32_t k1 = (u >> 10) & 0x7FFu;
      if (k1 > sel1)       atomicOr(&bm[j >> 5],  1u << (j & 31));
      else if (k1 == sel1) atomicOr(&tbm[j >> 5], 1u << (j & 31));
    }
  }
  __syncthreads();
  if (tid == 0) {
    unsigned int run = 0;
    for (int w = 0; w < 512; ++w) { pre[w] = run; run += __popc(tbm[w]); }
  }
  __syncthreads();
  for (int j = tid; j < D_SAE; j += 256) {
    unsigned int wm = tbm[j >> 5];
    if (wm & (1u << (j & 31))) {
      unsigned int rank = pre[j >> 5] + __popc(wm & ((1u << (j & 31)) - 1u));
      if (rank < nTie) atomicOr(&bm[j >> 5], 1u << (j & 31));
    }
  }
  __syncthreads();
  if (tid == 0) {
    unsigned int run = 0;
    for (int w = 0; w < 512; ++w) { pre[w] = run; run += __popc(bm[w]); }
  }
  __syncthreads();
  float* vrow = vals + (size_t)row * KTOP;
  int*   irow = idxs + (size_t)row * KTOP;
  for (int j = tid; j < D_SAE; j += 256) {
    unsigned int wm = bm[j >> 5];
    if (wm & (1u << (j & 31))) {
      unsigned int pos = pre[j >> 5] + __popc(wm & ((1u << (j & 31)) - 1u));
      vrow[pos] = p[j];
      irow[pos] = j;
    }
  }
}

// ---------------------------------------------------------------------------
// Sparse decode: x_hat[row] = b_dec + sum_j vals[j] * W_dec[idx[j], :]
// ---------------------------------------------------------------------------
__global__ __launch_bounds__(256) void decode_kernel(
    const float* __restrict__ vals, const int* __restrict__ idxs,
    const float* __restrict__ W_dec, const float* __restrict__ b_dec,
    float* __restrict__ out)
{
  __shared__ float sv[KTOP];
  __shared__ int   si[KTOP];
  const int row = blockIdx.x;
  const int tid = threadIdx.x;
  if (tid < KTOP) {
    sv[tid] = vals[(size_t)row * KTOP + tid];
    si[tid] = idxs[(size_t)row * KTOP + tid];
  }
  __syncthreads();
  const int c0 = tid * 8;
  float4 a0 = *(const float4*)(b_dec + c0);
  float4 a1 = *(const float4*)(b_dec + c0 + 4);
  for (int j = 0; j < KTOP; ++j) {
    const float  s = sv[j];
    const float* w = W_dec + (size_t)si[j] * D_IN + c0;
    float4 w0 = *(const float4*)(w);
    float4 w1 = *(const float4*)(w + 4);
    a0.x = fmaf(s, w0.x, a0.x); a0.y = fmaf(s, w0.y, a0.y);
    a0.z = fmaf(s, w0.z, a0.z); a0.w = fmaf(s, w0.w, a0.w);
    a1.x = fmaf(s, w1.x, a1.x); a1.y = fmaf(s, w1.y, a1.y);
    a1.z = fmaf(s, w1.z, a1.z); a1.w = fmaf(s, w1.w, a1.w);
  }
  float* o = out + (size_t)row * D_IN + c0;
  *(float4*)o       = a0;
  *(float4*)(o + 4) = a1;
}

// ---------------------------------------------------------------------------
// Launch. Workspace layout (assumes ws_size >= ~420 MB; device has 432 GiB):
//   Ahi/Alo: 16 MB each | Whi/Wlo: 64 MB each | post: 256 MB | vals/idx: 1.6 MB each
// ---------------------------------------------------------------------------
extern "C" void kernel_launch(void* const* d_in, const int* in_sizes, int n_in,
                              void* d_out, int out_size, void* d_ws, size_t ws_size,
                              hipStream_t stream)
{
  (void)in_sizes; (void)n_in; (void)out_size; (void)ws_size;
  const float* x     = (const float*)d_in[0];
  const float* W_enc = (const float*)d_in[1];
  const float* b_enc = (const float*)d_in[2];
  const float* W_dec = (const float*)d_in[3];
  const float* b_dec = (const float*)d_in[4];

  char* ws = (char*)d_ws;
  size_t off = 0;
  uint16_t* Ahi  = (uint16_t*)(ws + off); off += (size_t)BATCH * D_IN * 2;
  uint16_t* Alo  = (uint16_t*)(ws + off); off += (size_t)BATCH * D_IN * 2;
  uint16_t* Whi  = (uint16_t*)(ws + off); off += (size_t)D_SAE * D_IN * 2;
  uint16_t* Wlo  = (uint16_t*)(ws + off); off += (size_t)D_SAE * D_IN * 2;
  float*    post = (float*)(ws + off);    off += (size_t)BATCH * D_SAE * 4;
  float*    vals = (float*)(ws + off);    off += (size_t)BATCH * KTOP * 4;
  int*      idxs = (int*)(ws + off);      off += (size_t)BATCH * KTOP * 4;

  prep_x_kernel<<<(BATCH * D_IN / 4 + 255) / 256, 256, 0, stream>>>(x, b_dec, Ahi, Alo);
  prep_w_kernel<<<dim3(D_SAE / 32, D_IN / 32), dim3(32, 8), 0, stream>>>(W_enc, Whi, Wlo);
  encode_gemm_kernel<<<dim3(D_SAE / BN, BATCH / BM), 256, 0, stream>>>(
      Ahi, Alo, Whi, Wlo, b_enc, post);
  topk_kernel<<<BATCH, 256, 0, stream>>>(post, vals, idxs);
  decode_kernel<<<BATCH, 256, 0, stream>>>(vals, idxs, W_dec, b_dec, (float*)d_out);
}
